// Model_39676907884557
// MI455X (gfx1250) — matrix operand order repaired and checked by reading, not compiled, not run on hardware
//
#include <hip/hip_runtime.h>
#include <stdint.h>

// ---------------------------------------------------------------------------
// CDNA5 (gfx1250) wave32 WMMA attention pipeline, LDS-tiled with async copies.
// Matmuls: v_wmma_f32_16x16x32_bf16; staging: global_load_async_to_lds_b128.
// ---------------------------------------------------------------------------

typedef __attribute__((ext_vector_type(16))) __bf16 v16bf;
typedef __attribute__((ext_vector_type(2)))  __bf16 v2bf;
typedef __attribute__((ext_vector_type(8)))  float  v8f;

union FragBF { v16bf v; uint32_t u[8]; uint4 q[2]; };
union AccF   { v8f   v; float    f[8]; };

__device__ __forceinline__ uint32_t pack2_bf16(float lo, float hi) {
#if __has_builtin(__builtin_amdgcn_cvt_pk_bf16_f32)
  union { v2bf v; uint32_t u; } r;
  r.v = __builtin_amdgcn_cvt_pk_bf16_f32(lo, hi);
  return r.u;
#else
  uint32_t a = __float_as_uint(lo);
  uint32_t b = __float_as_uint(hi);
  a = (a + 0x7FFFu + ((a >> 16) & 1u)) >> 16;   // RNE
  b = (b + 0x7FFFu + ((b >> 16) & 1u)) >> 16;
  return (a & 0xFFFFu) | (b << 16);
#endif
}

__device__ __forceinline__ uint16_t f32_to_bf16(float x) {
  uint32_t a = __float_as_uint(x);
  return (uint16_t)((a + 0x7FFFu + ((a >> 16) & 1u)) >> 16);
}

// 16-byte async copy global -> LDS (per-lane). ldsOff = wave-relative LDS byte
// address (low 32 bits of the flat pointer of a __shared__ object).
__device__ __forceinline__ void async_cp16(uint32_t ldsOff, uint64_t gaddr) {
  asm volatile("global_load_async_to_lds_b128 %0, %1, off"
               :: "v"(ldsOff), "v"(gaddr) : "memory");
}
__device__ __forceinline__ void wait_async0() {
  asm volatile("s_wait_asynccnt 0x0" ::: "memory");
}

// ---------------------------------------------------------------------------
// fp32 -> bf16 bulk conversion (memory-bound one-shot pass).
// ---------------------------------------------------------------------------
__global__ __launch_bounds__(256) void cvt_f32_bf16(const float* __restrict__ in,
                                                    uint16_t* __restrict__ out,
                                                    size_t n8) {
  size_t i = (size_t)blockIdx.x * 256 + threadIdx.x;
  if (i >= n8) return;
  float4 x = ((const float4*)in)[i * 2];
  float4 y = ((const float4*)in)[i * 2 + 1];
  uint4 o;
  o.x = pack2_bf16(x.x, x.y); o.y = pack2_bf16(x.z, x.w);
  o.z = pack2_bf16(y.x, y.y); o.w = pack2_bf16(y.z, y.w);
  ((uint4*)out)[i] = o;
}

// ---------------------------------------------------------------------------
// NT GEMM: C[M,N] = alpha * A[M,K] * B[N,K]^T (+ bias row/col), bf16 in.
// Block = 256 threads = 8 waves (4M x 2N); block tile 128M x 128N; K-step 32.
// Wave tile 32M x 64N (2x4 accumulators -> 8 WMMAs / 12 ds_read_b128 per step).
// Double-buffered LDS staging via per-lane async copies.
// batch strides in BYTES, lda/ldb/ldc in elements.
// ---------------------------------------------------------------------------
#define ROWP 40  // padded LDS row stride in elements (80B -> conflict-free b128)

template<int BIAS, bool OUT_BF16>
__global__ __launch_bounds__(256) void gemm_nt_wmma(
    const void* __restrict__ Ap, size_t strideA, int lda,
    const void* __restrict__ Bp, size_t strideB, int ldb,
    const float* __restrict__ bias,
    void* __restrict__ Cp, size_t strideC, int ldc,
    int K, float alpha)
{
  __shared__ __align__(16) uint16_t As[2][128 * ROWP];
  __shared__ __align__(16) uint16_t Bs[2][128 * ROWP];

  const int t    = threadIdx.x;
  const int lane = t & 31;
  const int wave = t >> 5;
  const int r    = lane & 15;
  const int half = lane >> 4;
  const int wm   = (wave >> 1) * 32;   // wave M offset in block tile (0..96)
  const int wn   = (wave & 1) * 64;    // wave N offset in block tile (0/64)
  const int Mblk = blockIdx.y * 128;
  const int Nblk = blockIdx.x * 128;

  const uint16_t* Ab = (const uint16_t*)((const char*)Ap + (size_t)blockIdx.z * strideA);
  const uint16_t* Bb = (const uint16_t*)((const char*)Bp + (size_t)blockIdx.z * strideB);
  char*           Cb = (char*)Cp + (size_t)blockIdx.z * strideC;

  // Staging: A and B tiles are each 128x32 bf16 = 8KB -> 2 chunks of 16B per
  // thread per tile (rows srow and srow+64).
  const int srow = t >> 2;             // 0..63
  const int scol = (t & 3) * 8;        // element offset 0,8,16,24
  uint64_t gA0 = (uint64_t)(Ab + (size_t)(Mblk + srow) * lda + scol);
  uint64_t gA1 = (uint64_t)(Ab + (size_t)(Mblk + srow + 64) * lda + scol);
  uint64_t gB0 = (uint64_t)(Bb + (size_t)(Nblk + srow) * ldb + scol);
  uint64_t gB1 = (uint64_t)(Bb + (size_t)(Nblk + srow + 64) * ldb + scol);

  uint32_t lA0[2], lA1[2], lB0[2], lB1[2];
#pragma unroll
  for (int bb = 0; bb < 2; ++bb) {
    lA0[bb] = (uint32_t)(uintptr_t)&As[bb][srow * ROWP + scol];
    lA1[bb] = (uint32_t)(uintptr_t)&As[bb][(srow + 64) * ROWP + scol];
    lB0[bb] = (uint32_t)(uintptr_t)&Bs[bb][srow * ROWP + scol];
    lB1[bb] = (uint32_t)(uintptr_t)&Bs[bb][(srow + 64) * ROWP + scol];
  }

  // Prologue: stage k-step 0 into buffer 0.
  async_cp16(lA0[0], gA0);
  async_cp16(lA1[0], gA1);
  async_cp16(lB0[0], gB0);
  async_cp16(lB1[0], gB1);
  gA0 += 64; gA1 += 64; gB0 += 64; gB1 += 64;   // advance 32 bf16 elements

  AccF acc[2][4];
#pragma unroll
  for (int i = 0; i < 2; ++i)
#pragma unroll
    for (int j = 0; j < 4; ++j)
#pragma unroll
      for (int e = 0; e < 8; ++e) acc[i][j].f[e] = 0.0f;

  const int nIter = K >> 5;
  for (int it = 0; it < nIter; ++it) {
    const int cur = it & 1;
    wait_async0();
    __syncthreads();                   // current buffer filled by all waves

    if (it + 1 < nIter) {
      const int nxt = cur ^ 1;
      async_cp16(lA0[nxt], gA0);
      async_cp16(lA1[nxt], gA1);
      async_cp16(lB0[nxt], gB0);
      async_cp16(lB1[nxt], gB1);
      gA0 += 64; gA1 += 64; gB0 += 64; gB1 += 64;
    }

    // Fragment loads from LDS (layouts per ISA 7.12.2):
    //  A 16x32: VGPR0-3 K=8*half+[0..7], VGPR4-7 K=16+8*half+[0..7]
    //  B 32x16: VGPR0-3 K=8*half+[0..7], VGPR4-7 K=16+8*half+[0..7]
    FragBF a[2], b[4];
#pragma unroll
    for (int i = 0; i < 2; ++i) {
      const uint16_t* as = &As[cur][(wm + i * 16 + r) * ROWP + 8 * half];
      a[i].q[0] = *(const uint4*)as;
      a[i].q[1] = *(const uint4*)(as + 16);
    }
#pragma unroll
    for (int j = 0; j < 4; ++j) {
      const uint16_t* bs = &Bs[cur][(wn + j * 16 + r) * ROWP + 8 * half];
      b[j].q[0] = *(const uint4*)bs;
      b[j].q[1] = *(const uint4*)(bs + 16);
    }
#pragma unroll
    for (int i = 0; i < 2; ++i)
#pragma unroll
      for (int j = 0; j < 4; ++j)
        acc[i][j].v = __builtin_amdgcn_wmma_f32_16x16x32_bf16(
            false, a[i].v, false, b[j].v, (short)0, acc[i][j].v, false, false);

    __syncthreads();                   // all waves done reading `cur`
  }

  // Epilogue. C layout: VGPR j -> row = j + 8*half, col = lane&15.
#pragma unroll
  for (int i = 0; i < 2; ++i) {
#pragma unroll
    for (int jn = 0; jn < 4; ++jn) {
      const int col = Nblk + wn + jn * 16 + r;
#pragma unroll
      for (int j = 0; j < 8; ++j) {
        const int row = Mblk + wm + i * 16 + j + 8 * half;
        float v = acc[i][jn].f[j] * alpha;
        if (BIAS == 1) v += bias[row];
        if (BIAS == 2) v += bias[col];
        if (OUT_BF16) ((uint16_t*)Cb)[(size_t)row * ldc + col] = f32_to_bf16(v);
        else          ((float*)Cb)[(size_t)row * ldc + col] = v;
      }
    }
  }
}

// ---------------------------------------------------------------------------
// Row softmax over 2048 fp32, writing bf16 IN PLACE (row stride stays 8192B,
// i.e. 4096 bf16 elements). One block/row, 256 threads x 8 elements.
// ---------------------------------------------------------------------------
__global__ __launch_bounds__(256) void softmax_bf16_inplace(float* __restrict__ scores)
{
  float* p = scores + (size_t)blockIdx.x * 2048;
  const int t = threadIdx.x;

  float4 x0 = ((const float4*)p)[t * 2];
  float4 x1 = ((const float4*)p)[t * 2 + 1];
  float v[8] = {x0.x, x0.y, x0.z, x0.w, x1.x, x1.y, x1.z, x1.w};

  __shared__ float red[256];

  float m = v[0];
#pragma unroll
  for (int i = 1; i < 8; ++i) m = fmaxf(m, v[i]);
  red[t] = m;
  __syncthreads();
  for (int s = 128; s > 0; s >>= 1) {
    if (t < s) red[t] = fmaxf(red[t], red[t + s]);
    __syncthreads();
  }
  m = red[0];
  __syncthreads();

  float sum = 0.0f;
#pragma unroll
  for (int i = 0; i < 8; ++i) { v[i] = __expf(v[i] - m); sum += v[i]; }
  red[t] = sum;
  __syncthreads();
  for (int s = 128; s > 0; s >>= 1) {
    if (t < s) red[t] += red[t + s];
    __syncthreads();
  }
  const float inv = 1.0f / red[0];
  __syncthreads();   // all fp32 reads done before aliasing bf16 writes

  uint4 o;
  o.x = pack2_bf16(v[0] * inv, v[1] * inv);
  o.y = pack2_bf16(v[2] * inv, v[3] * inv);
  o.z = pack2_bf16(v[4] * inv, v[5] * inv);
  o.w = pack2_bf16(v[6] * inv, v[7] * inv);
  ((uint4*)p)[t] = o;
}

// ---------------------------------------------------------------------------
// Orchestration. B=8, S=H=1024, NK=2048.
// Workspace (bytes):
//   [  0,  16M) qbf  bf16 query            [ 16M, 48M) kbf  bf16 key
//   [ 48M, 50M) wqb  bf16 Wq               [ 50M, 52M) wkb  bf16 Wk
//   [ 52M, 54M) wvb  bf16 Wv
//   [ 54M, 70M) Qt   bf16 (B,H,S)   = Wq * query^T + bq
//   [ 70M,102M) Kp   bf16 (B,NK,H)  = key * Wk^T + bk
//   [102M,134M) Vp   bf16 (B,NK,H)  = key * Wv^T + bv
//   [134M,198M) sc   f32  (B,H,NK)  -> softmaxed bf16 in place (lda 4096)
// ---------------------------------------------------------------------------
extern "C" void kernel_launch(void* const* d_in, const int* in_sizes, int n_in,
                              void* d_out, int out_size, void* d_ws, size_t ws_size,
                              hipStream_t stream) {
  (void)in_sizes; (void)n_in; (void)ws_size;

  const float* query = (const float*)d_in[0];
  const float* key   = (const float*)d_in[1];
  const float* Wq    = (const float*)d_in[5];
  const float* bq    = (const float*)d_in[6];
  const float* Wk    = (const float*)d_in[7];
  const float* bk    = (const float*)d_in[8];
  const float* Wv    = (const float*)d_in[9];
  const float* bv    = (const float*)d_in[10];

  char* ws = (char*)d_ws;
  const size_t MB = (size_t)1024 * 1024;
  uint16_t* qbf = (uint16_t*)(ws);
  uint16_t* kbf = (uint16_t*)(ws + 16 * MB);
  uint16_t* wqb = (uint16_t*)(ws + 48 * MB);
  uint16_t* wkb = (uint16_t*)(ws + 50 * MB);
  uint16_t* wvb = (uint16_t*)(ws + 52 * MB);
  void*     qt  = (void*)(ws + 54 * MB);
  void*     kp  = (void*)(ws + 70 * MB);
  void*     vp  = (void*)(ws + 102 * MB);
  float*    sc  = (float*)(ws + 134 * MB);

  const dim3 blk(256, 1, 1);

  // 0) one-shot fp32 -> bf16 conversions
  cvt_f32_bf16<<<dim3(4096), blk, 0, stream>>>(query, qbf, (size_t)1048576);
  cvt_f32_bf16<<<dim3(8192), blk, 0, stream>>>(key,   kbf, (size_t)2097152);
  cvt_f32_bf16<<<dim3(512),  blk, 0, stream>>>(Wq,    wqb, (size_t)131072);
  cvt_f32_bf16<<<dim3(512),  blk, 0, stream>>>(Wk,    wkb, (size_t)131072);
  cvt_f32_bf16<<<dim3(512),  blk, 0, stream>>>(Wv,    wvb, (size_t)131072);

  const size_t sQbf = 1024 * 1024 * 2;   // query bf16 batch stride (bytes)
  const size_t sKbf = 2048 * 1024 * 2;   // key bf16 batch stride
  const size_t sQt  = 1024 * 1024 * 2;
  const size_t sKP  = 2048 * 1024 * 2;
  const size_t sSc  = 1024 * 2048 * 4;
  const size_t sO   = 1024 * 1024 * 4;

  // 1) Qt = Wq * query^T + bq  (H x S), bias per row
  gemm_nt_wmma<1, true><<<dim3(8, 8, 8), blk, 0, stream>>>(
      wqb, 0, 1024, qbf, sQbf, 1024, bq, qt, sQt, 1024, 1024, 1.0f);

  // 2) Kp = key * Wk^T + bk  (NK x H), bias per col
  gemm_nt_wmma<2, true><<<dim3(8, 16, 8), blk, 0, stream>>>(
      kbf, sKbf, 1024, wkb, 0, 1024, bk, kp, sKP, 1024, 1024, 1.0f);

  // 3) Vp = key * Wv^T + bv  (NK x H), bias per col
  gemm_nt_wmma<2, true><<<dim3(8, 16, 8), blk, 0, stream>>>(
      kbf, sKbf, 1024, wvb, 0, 1024, bv, vp, sKP, 1024, 1024, 1.0f);

  // 4) sc = (1/32) * Qt * Kp^T  (H x NK), f32
  gemm_nt_wmma<0, false><<<dim3(16, 8, 8), blk, 0, stream>>>(
      qt, sQt, 1024, kp, sKP, 1024, nullptr, sc, sSc, 2048, 1024, 0.03125f);

  // 5) softmax rows -> bf16 in place (row stride 4096 bf16 elements)
  softmax_bf16_inplace<<<dim3(8 * 1024), blk, 0, stream>>>(sc);

  // 6) out = attn * Vview^T : Vp viewed flat as (H x NK), ld = 2048
  gemm_nt_wmma<0, false><<<dim3(8, 8, 8), blk, 0, stream>>>(
      sc, sSc, 4096, vp, sKP, 2048, nullptr, d_out, sO, 1024, 2048, 1.0f);

  // 7) mask outputs are all-False -> zeros
  const size_t outElems = (size_t)8 * 1024 * 1024;
  if ((size_t)out_size > outElems) {
    hipMemsetAsync((char*)d_out + outElems * 4, 0,
                   ((size_t)out_size - outElems) * 4, stream);
  }
}
